// Decoder_45973329937301
// MI455X (gfx1250) — matrix operand order repaired and checked by reading, not compiled, not run on hardware
//
#include <hip/hip_runtime.h>
#include <cstddef>

// ---------------------------------------------------------------------------
// Types for CDNA5 WMMA (wave32): v_wmma_f32_16x16x32_bf16
// ---------------------------------------------------------------------------
typedef __bf16 bf16_t;
typedef __attribute__((ext_vector_type(16))) __bf16 v16bf;
typedef __attribute__((ext_vector_type(8)))  __bf16 v8bf;
typedef __attribute__((ext_vector_type(8)))  float  v8f;

// Problem constants (from the reference)
#define BB 64
#define PP 196
#define DD 2048
#define AA 512
#define HH 512
#define EE 512
#define VV 10000
#define LL 20
#define TT 19
#define XW (EE + DD)      // 2560: [emb | awe] concat width, also dec_w|f_beta_w pack

__device__ __forceinline__ float sigf(float x) { return 1.0f / (1.0f + __expf(-x)); }

// ---------------------------------------------------------------------------
// Register-blocked WMMA GEMM:
//   out[M,N] = X[M,K](bf16, ldx) * W[N,K](bf16, ldw)^T + bias[N]
// One wave computes a (16*MT) x (16*NT) tile: MT A-fragments, NT B-fragments,
// MT*NT accumulators -> each loaded fragment feeds MT (or NT) WMMAs.
//   acc  : accumulate into existing out
//   lens : if non-null, zero rows where t >= lens[row]-1 (masked logits)
// A per-lane layout (16-bit 16x32): row M=lane&15, 16 contiguous K at
// k0+16*(lane>>4).  B per-lane: column N=lane&15, 16 contiguous K at
// k0+16*(lane>>4) -> one 32B load.
// ---------------------------------------------------------------------------
template<int MT, int NT>
__global__ void __launch_bounds__(128)
gemm_bf16_kernel(float* __restrict__ out,
                 const bf16_t* __restrict__ X,
                 const bf16_t* __restrict__ W,
                 const float* __restrict__ bias,
                 int M, int N, int K,
                 int ldx, int ldw, int ldout,
                 int acc, const int* __restrict__ lens, int t)
{
    const int lane   = threadIdx.x & 31;
    const int wave   = blockIdx.x * (blockDim.x >> 5) + (threadIdx.x >> 5);
    const int tilesN = N / (16 * NT);
    const int tilesM = M / (16 * MT);
    if (wave >= tilesM * tilesN) return;

    const int tm   = wave / tilesN;
    const int tn   = wave - tm * tilesN;
    const int m0   = tm * 16 * MT;
    const int n0   = tn * 16 * NT;
    const int half = lane >> 4;
    const int l15  = lane & 15;

    const bf16_t* Xr[MT];
    const bf16_t* Wr[NT];
#pragma unroll
    for (int i = 0; i < MT; ++i)
        Xr[i] = X + (size_t)(m0 + i * 16 + l15) * ldx;
#pragma unroll
    for (int j = 0; j < NT; ++j)
        Wr[j] = W + (size_t)(n0 + j * 16 + l15) * ldw + 16 * half;

    v8f zero = {};
    v8f c[MT][NT];
#pragma unroll
    for (int i = 0; i < MT; ++i)
#pragma unroll
        for (int j = 0; j < NT; ++j) c[i][j] = zero;

    if (acc) {
#pragma unroll
        for (int i = 0; i < MT; ++i)
#pragma unroll
            for (int j = 0; j < NT; ++j)
#pragma unroll
                for (int r = 0; r < 8; ++r)
                    c[i][j][r] = out[(size_t)(m0 + i * 16 + r + 8 * half) * ldout
                                     + n0 + j * 16 + l15];
    }

    for (int k0 = 0; k0 < K; k0 += 32) {
        v16bf a[MT];
#pragma unroll
        for (int i = 0; i < MT; ++i) {
            a[i] = *(const v16bf*)(Xr[i] + k0 + 16 * half);
        }
        if constexpr (NT == 1) {
            // weight-streaming regime: pull the next chunks of this B row
            // toward the WGP ahead of use (global_prefetch_b8)
            __builtin_prefetch(Wr[0] + k0 + 1024, 0, 1);
        }
#pragma unroll
        for (int j = 0; j < NT; ++j) {
            v16bf b = *(const v16bf*)(Wr[j] + k0);
#pragma unroll
            for (int i = 0; i < MT; ++i)
                c[i][j] = __builtin_amdgcn_wmma_f32_16x16x32_bf16(
                    /*neg_a=*/false, a[i], /*neg_b=*/false, b,
                    /*c_mod=*/(short)0, c[i][j],
                    /*reuse_a=*/false, /*reuse_b=*/false);
        }
    }

#pragma unroll
    for (int j = 0; j < NT; ++j) {
        const int ncol = n0 + j * 16 + l15;
        const float bv = bias ? bias[ncol] : 0.0f;
#pragma unroll
        for (int i = 0; i < MT; ++i) {
#pragma unroll
            for (int r = 0; r < 8; ++r) {
                const int row = m0 + i * 16 + r + 8 * half;
                float v = c[i][j][r] + bv;
                if (lens && t >= lens[row] - 1) v = 0.0f;
                out[(size_t)row * ldout + ncol] = v;
            }
        }
    }
}

// ---------------------------------------------------------------------------
// f32 -> bf16 convert / f32 copy (grid-stride)
// ---------------------------------------------------------------------------
__global__ void cvt_bf16_kernel(bf16_t* __restrict__ dst,
                                const float* __restrict__ src, int n)
{
    for (int i = blockIdx.x * blockDim.x + threadIdx.x; i < n;
         i += gridDim.x * blockDim.x)
        dst[i] = (bf16_t)src[i];
}

__global__ void copy_f32_kernel(float* __restrict__ dst,
                                const float* __restrict__ src, int n)
{
    for (int i = blockIdx.x * blockDim.x + threadIdx.x; i < n;
         i += gridDim.x * blockDim.x)
        dst[i] = src[i];
}

// ---------------------------------------------------------------------------
// mean over P of features: meanf_bf[b,d]
// ---------------------------------------------------------------------------
__global__ void mean_feats_kernel(const float* __restrict__ feats,
                                  bf16_t* __restrict__ meanf_bf)
{
    const int idx = blockIdx.x * blockDim.x + threadIdx.x;   // BB*DD
    if (idx >= BB * DD) return;
    const int b = idx >> 11, d = idx & (DD - 1);
    const float* f = feats + (size_t)b * PP * DD + d;
    float s = 0.0f;
    for (int p = 0; p < PP; ++p) s += f[(size_t)p * DD];
    meanf_bf[idx] = (bf16_t)(s * (1.0f / (float)PP));
}

// ---------------------------------------------------------------------------
// embedding gather -> bf16, directly into per-step xcat buffers:
//   xcat_all[t][b][0:E] = emb[captions[b][t]]
// ---------------------------------------------------------------------------
__global__ void gather_emb_kernel(const float* __restrict__ emb,
                                  const int* __restrict__ captions,
                                  bf16_t* __restrict__ xcat_all)
{
    const int idx = blockIdx.x * blockDim.x + threadIdx.x;   // BB*TT*EE
    if (idx >= BB * TT * EE) return;
    const int e  = idx & (EE - 1);
    const int bt = idx >> 9;
    const int b  = bt / TT;
    const int t  = bt - b * TT;
    const int cap = captions[b * LL + t];
    xcat_all[((size_t)t * BB + b) * XW + e] = (bf16_t)emb[(size_t)cap * EE + e];
}

// ---------------------------------------------------------------------------
// Fused per-step attention:
//   e[b,p] = sum_a relu(att1[b,p,a] + att2[b,a]) * att_w[a] + att_b
//   alpha  = softmax_p(e)        (masked copy -> d_out alpha region)
//   awe[b,d] = sigmoid(fbl[b,d]) * sum_p alpha[b,p] * feats[b,p,d]
//              -> xcat_t[b][E+d]  (bf16)
// comb = [att2 | fbl] per batch row (width XW).  One block per batch element.
// ---------------------------------------------------------------------------
__global__ void __launch_bounds__(256)
attn_step_kernel(const float* __restrict__ att1,
                 const float* __restrict__ comb,
                 const float* __restrict__ att_w,
                 const float* __restrict__ att_b,
                 const bf16_t* __restrict__ feats_bf,
                 bf16_t* __restrict__ xcat_t,
                 float* __restrict__ out_alpha,
                 const int* __restrict__ lens, int t)
{
    __shared__ float s_att2[AA];
    __shared__ float s_w[AA];
    __shared__ float s_red[256];
    __shared__ float s_al[PP];

    const int b   = blockIdx.x;
    const int tid = threadIdx.x;
    for (int a = tid; a < AA; a += 256) {
        s_att2[a] = comb[(size_t)b * XW + a];
        s_w[a]    = att_w[a];
    }
    __syncthreads();

    float e = -1e30f;
    if (tid < PP) {
        const float* row = att1 + ((size_t)b * PP + tid) * AA;
        float s = att_b[0];
        for (int a = 0; a < AA; a += 4) {
            const float4 v = *(const float4*)(row + a);
            s += fmaxf(v.x + s_att2[a + 0], 0.0f) * s_w[a + 0]
               + fmaxf(v.y + s_att2[a + 1], 0.0f) * s_w[a + 1]
               + fmaxf(v.z + s_att2[a + 2], 0.0f) * s_w[a + 2]
               + fmaxf(v.w + s_att2[a + 3], 0.0f) * s_w[a + 3];
        }
        e = s;
    }

    s_red[tid] = e;
    __syncthreads();
    for (int off = 128; off > 0; off >>= 1) {
        if (tid < off) s_red[tid] = fmaxf(s_red[tid], s_red[tid + off]);
        __syncthreads();
    }
    const float mx = s_red[0];
    __syncthreads();

    const float ex = (tid < PP) ? __expf(e - mx) : 0.0f;
    s_red[tid] = ex;
    __syncthreads();
    for (int off = 128; off > 0; off >>= 1) {
        if (tid < off) s_red[tid] += s_red[tid + off];
        __syncthreads();
    }
    const float inv = 1.0f / s_red[0];

    if (tid < PP) {
        const float al = ex * inv;
        s_al[tid] = al;
        const int active = (t < lens[b] - 1);
        out_alpha[((size_t)b * TT + t) * PP + tid] = active ? al : 0.0f;
    }
    __syncthreads();

    // gated attention-weighted encoding -> xcat awe half
    for (int d = tid; d < DD; d += 256) {
        const bf16_t* f = feats_bf + (size_t)b * PP * DD + d;
        float s = 0.0f;
        for (int p = 0; p < PP; ++p) s += s_al[p] * (float)f[(size_t)p * DD];
        const float g = sigf(comb[(size_t)b * XW + EE + d]);
        xcat_t[(size_t)b * XW + EE + d] = (bf16_t)(g * s);
    }
}

// ---------------------------------------------------------------------------
// LSTM pointwise: gates(b,4H) order i,f,g,o -> c (f32, in place), h2 (bf16)
// ---------------------------------------------------------------------------
__global__ void lstm_update_kernel(const float* __restrict__ gates,
                                   float* __restrict__ c,
                                   bf16_t* __restrict__ h_bf)
{
    const int idx = blockIdx.x * blockDim.x + threadIdx.x;   // BB*HH
    if (idx >= BB * HH) return;
    const int b = idx >> 9, j = idx & (HH - 1);
    const float* g = gates + (size_t)b * 4 * HH;
    const float ig = sigf(g[j]);
    const float fg = sigf(g[HH + j]);
    const float gg = tanhf(g[2 * HH + j]);
    const float og = sigf(g[3 * HH + j]);
    const float c2 = fg * c[idx] + ig * gg;
    c[idx]    = c2;
    h_bf[idx] = (bf16_t)(og * tanhf(c2));
}

// ---------------------------------------------------------------------------
// Host side
// ---------------------------------------------------------------------------
template<int MT, int NT>
static inline void launch_gemm(hipStream_t s, float* out, const bf16_t* X,
                               const bf16_t* W, const float* bias,
                               int M, int N, int K, int ldx, int ldw, int ldout,
                               int acc, const int* lens, int t)
{
    const int tiles = (M / (16 * MT)) * (N / (16 * NT));
    gemm_bf16_kernel<MT, NT><<<dim3((tiles + 3) / 4), dim3(128), 0, s>>>(
        out, X, W, bias, M, N, K, ldx, ldw, ldout, acc, lens, t);
}

static inline void launch_cvt(hipStream_t s, bf16_t* dst, const float* src, int n)
{
    int blocks = (n + 255) / 256;
    if (blocks > 65535) blocks = 65535;
    cvt_bf16_kernel<<<dim3(blocks), dim3(256), 0, s>>>(dst, src, n);
}

extern "C" void kernel_launch(void* const* d_in, const int* in_sizes, int n_in,
                              void* d_out, int out_size, void* d_ws, size_t ws_size,
                              hipStream_t stream)
{
    (void)in_sizes; (void)n_in; (void)out_size; (void)ws_size;

    const float* features = (const float*)d_in[0];
    const int*   captions = (const int*)d_in[1];
    const int*   lengths  = (const int*)d_in[2];
    const float* emb      = (const float*)d_in[3];
    const float* W_ih     = (const float*)d_in[4];
    const float* b_ih     = (const float*)d_in[5];
    const float* W_hh     = (const float*)d_in[6];
    const float* b_hh     = (const float*)d_in[7];
    const float* h_fc_w   = (const float*)d_in[8];
    const float* h_fc_b   = (const float*)d_in[9];
    const float* c_fc_w   = (const float*)d_in[10];
    const float* c_fc_b   = (const float*)d_in[11];
    const float* f_beta_w = (const float*)d_in[12];
    const float* f_beta_b = (const float*)d_in[13];
    const float* cls_w    = (const float*)d_in[14];
    const float* cls_b    = (const float*)d_in[15];
    const float* enc_w    = (const float*)d_in[16];
    const float* enc_b    = (const float*)d_in[17];
    const float* dec_w    = (const float*)d_in[18];
    const float* dec_b    = (const float*)d_in[19];
    const float* att_w    = (const float*)d_in[20];
    const float* att_b    = (const float*)d_in[21];

    float* y_out     = (float*)d_out;                       // (B,T,V)
    float* alpha_out = y_out + (size_t)BB * TT * VV;        // (B,T,P)

    // -------- workspace carve-up (256B aligned) --------
    char*  w   = (char*)d_ws;
    size_t off = 0;
    auto alloc = [&](size_t bytes) -> void* {
        void* p = w + off;
        off += (bytes + 255) & ~(size_t)255;
        return p;
    };
    bf16_t* feats_bf = (bf16_t*)alloc((size_t)BB * PP * DD * 2);
    bf16_t* encw_bf  = (bf16_t*)alloc((size_t)AA * DD * 2);
    bf16_t* Wih_bf   = (bf16_t*)alloc((size_t)4 * HH * XW * 2);
    bf16_t* Whh_bf   = (bf16_t*)alloc((size_t)4 * HH * HH * 2);
    bf16_t* cls_bf   = (bf16_t*)alloc((size_t)VV * HH * 2);
    bf16_t* wcomb_bf = (bf16_t*)alloc((size_t)XW * HH * 2);   // [dec_w ; f_beta_w]
    bf16_t* hfc_bf   = (bf16_t*)alloc((size_t)HH * DD * 2);
    bf16_t* cfc_bf   = (bf16_t*)alloc((size_t)HH * DD * 2);
    bf16_t* meanf_bf = (bf16_t*)alloc((size_t)BB * DD * 2);
    bf16_t* xcat_all = (bf16_t*)alloc((size_t)TT * BB * XW * 2);
    bf16_t* h_bf     = (bf16_t*)alloc((size_t)BB * HH * 2);
    float*  att1     = (float*)alloc((size_t)BB * PP * AA * 4);
    float*  h_f      = (float*)alloc((size_t)BB * HH * 4);
    float*  c_f      = (float*)alloc((size_t)BB * HH * 4);
    float*  comb     = (float*)alloc((size_t)BB * XW * 4);    // [att2 | fbl]
    float*  bcomb    = (float*)alloc((size_t)XW * 4);         // [dec_b | f_beta_b]
    float*  gates    = (float*)alloc((size_t)BB * 4 * HH * 4);

    // -------- time-invariant setup --------
    launch_cvt(stream, feats_bf, features, BB * PP * DD);
    launch_cvt(stream, encw_bf,  enc_w,    AA * DD);
    launch_cvt(stream, Wih_bf,   W_ih,     4 * HH * XW);
    launch_cvt(stream, Whh_bf,   W_hh,     4 * HH * HH);
    launch_cvt(stream, cls_bf,   cls_w,    VV * HH);
    launch_cvt(stream, wcomb_bf,            dec_w,    AA * HH);
    launch_cvt(stream, wcomb_bf + AA * HH,  f_beta_w, DD * HH);
    launch_cvt(stream, hfc_bf,   h_fc_w,   HH * DD);
    launch_cvt(stream, cfc_bf,   c_fc_w,   HH * DD);
    copy_f32_kernel<<<dim3(2), dim3(256), 0, stream>>>(bcomb, dec_b, AA);
    copy_f32_kernel<<<dim3(8), dim3(256), 0, stream>>>(bcomb + AA, f_beta_b, DD);

    gather_emb_kernel<<<dim3((BB * TT * EE + 255) / 256), dim3(256), 0, stream>>>(
        emb, captions, xcat_all);
    mean_feats_kernel<<<dim3((BB * DD + 255) / 256), dim3(256), 0, stream>>>(
        features, meanf_bf);

    // h0 = mean_f @ h_fc_w^T + b ; c0 likewise
    launch_gemm<4, 1>(stream, h_f, meanf_bf, hfc_bf, h_fc_b,
                      BB, HH, DD, DD, DD, HH, 0, nullptr, 0);
    launch_gemm<4, 1>(stream, c_f, meanf_bf, cfc_bf, c_fc_b,
                      BB, HH, DD, DD, DD, HH, 0, nullptr, 0);
    launch_cvt(stream, h_bf, h_f, BB * HH);

    // att1 = feats @ enc_w^T + enc_b  (hoisted: 12544 x 512 x 2048, 32x64 tiles)
    launch_gemm<2, 4>(stream, att1, feats_bf, encw_bf, enc_b,
                      BB * PP, AA, DD, DD, DD, AA, 0, nullptr, 0);

    // -------- sequential decode --------
    for (int t = 0; t < TT; ++t) {
        bf16_t* xcat_t = xcat_all + (size_t)t * BB * XW;
        // [att2 | f_beta logits] = h @ [dec_w ; f_beta_w]^T + [dec_b ; f_beta_b]
        launch_gemm<4, 1>(stream, comb, h_bf, wcomb_bf, bcomb,
                          BB, XW, HH, HH, HH, XW, 0, nullptr, 0);
        // attention scores + softmax + gated awe -> xcat_t[:, E:]
        attn_step_kernel<<<dim3(BB), dim3(256), 0, stream>>>(
            att1, comb, att_w, att_b, feats_bf, xcat_t, alpha_out, lengths, t);
        // gates = xcat @ W_ih^T + b_ih + h @ W_hh^T + b_hh
        launch_gemm<4, 1>(stream, gates, xcat_t, Wih_bf, b_ih,
                          BB, 4 * HH, XW, XW, XW, 4 * HH, 0, nullptr, 0);
        launch_gemm<4, 1>(stream, gates, h_bf, Whh_bf, b_hh,
                          BB, 4 * HH, HH, HH, HH, 4 * HH, 1, nullptr, 0);
        // LSTM pointwise -> c (in place), h2 (bf16)
        lstm_update_kernel<<<dim3((BB * HH + 255) / 256), dim3(256), 0, stream>>>(
            gates, c_f, h_bf);
        // y_t = h2 @ cls_w^T + cls_b, length-masked, straight into d_out
        launch_gemm<4, 1>(stream, y_out + (size_t)t * VV, h_bf, cls_bf, cls_b,
                          BB, VV, HH, HH, HH, TT * VV, 0, lengths, t);
    }
}
